// GPTTransformer_68667937129096
// MI455X (gfx1250) — matrix operand order repaired and checked by reading, not compiled, not run on hardware
//
#include <hip/hip_runtime.h>

typedef __attribute__((ext_vector_type(16))) _Float16 v16h;
typedef __attribute__((ext_vector_type(8)))  float    v8f;

#define Bsz 2
#define Sq  1024
#define Dm  512
#define Hh  8
#define Ll  6
#define FFd 2048
#define Vv  32000
#define HDd 64

// ---------------- fragment helpers (CDNA5 16-bit WMMA VGPR layouts) ----------
union F16Frag { uint4 u[2]; v16h h; };

// A fragment 16x32 (MxK): lane<16 row=lane, K = 0..7 / 16..23 ; lane>=16 K = 8..15 / 24..31
__device__ inline v16h ldsA(const _Float16* base, int pitch, int row0, int lane) {
  const _Float16* p = base + (row0 + (lane & 15)) * pitch;
  int hi8 = (lane >> 4) * 8;
  F16Frag f;
  f.u[0] = *(const uint4*)(p + hi8);
  f.u[1] = *(const uint4*)(p + 16 + hi8);
  return f.h;
}
// B fragment 32x16 (KxN) from [N][K] tile: lane<16 col=lane K=0..7 / 16..23 ; lane>=16 K=8..15 / 24..31
__device__ inline v16h ldsB(const _Float16* base, int pitch, int n0, int kc, int lane) {
  const _Float16* p = base + (n0 + (lane & 15)) * pitch + kc + (lane >> 4) * 8;
  F16Frag f;
  f.u[0] = *(const uint4*)(p);
  f.u[1] = *(const uint4*)(p + 16);
  return f.h;
}

// async copy: 32 contiguous bytes global -> LDS per lane (2 x b128, INST_OFFSET
// applies to both the global and LDS addresses per CDNA5 ISA 10.x async ops)
__device__ inline void async_copy32(unsigned lds_off, unsigned long long gaddr) {
  asm volatile(
      "global_load_async_to_lds_b128 %0, %1, off\n\t"
      "global_load_async_to_lds_b128 %0, %1, off offset:16"
      :: "v"(lds_off), "v"(gaddr) : "memory");
}
__device__ inline void wait_async(int n) {
  if (n == 0) asm volatile("s_wait_asynccnt 0" ::: "memory");
  else        asm volatile("s_wait_asynccnt 4" ::: "memory");
}

// ---------------- DPP16 intra-row (16-lane) butterfly reductions ------------
// xor1 = quad_perm(1,0,3,2)=0xB1 ; xor2 = quad_perm(2,3,0,1)=0x4E ;
// after those, each aligned 4-group is uniform, so row_half_mirror (0x141)
// and row_mirror (0x140) complete the 16-lane reduction. Pure VALU, no LDS.
__device__ inline float dpp_b1(float x) {
  return __uint_as_float(__builtin_amdgcn_update_dpp(0u, __float_as_uint(x), 0xB1, 0xf, 0xf, true));
}
__device__ inline float dpp_4e(float x) {
  return __uint_as_float(__builtin_amdgcn_update_dpp(0u, __float_as_uint(x), 0x4E, 0xf, 0xf, true));
}
__device__ inline float dpp_hm(float x) {
  return __uint_as_float(__builtin_amdgcn_update_dpp(0u, __float_as_uint(x), 0x141, 0xf, 0xf, true));
}
__device__ inline float dpp_mr(float x) {
  return __uint_as_float(__builtin_amdgcn_update_dpp(0u, __float_as_uint(x), 0x140, 0xf, 0xf, true));
}
__device__ inline float redmax16(float v) {
  v = fmaxf(v, dpp_b1(v));
  v = fmaxf(v, dpp_4e(v));
  v = fmaxf(v, dpp_hm(v));
  v = fmaxf(v, dpp_mr(v));
  return v;
}
__device__ inline float redsum16(float v) {
  v += dpp_b1(v);
  v += dpp_4e(v);
  v += dpp_hm(v);
  v += dpp_mr(v);
  return v;
}

// ---------------- generic WMMA GEMM --------------------------------------
// C[M,N] = A[M,K](f16) @ Wt[N,K](f16)^T + bias ; epilogue by MODE
// MODE 0: outh f16 row-major (+RELU)
// MODE 1: outh f16 scattered to [B,H,S,HD] (QKV)
// MODE 2: outf f32 = resid + result
// MODE 3: outf f32 row-major
#define BM 128
#define BN 128
#define BK 32
#define PA 40
#define PB 40

template <int MODE, bool RELU>
__global__ __launch_bounds__(256)
void gemm_kernel(const _Float16* __restrict__ A, const _Float16* __restrict__ Wt,
                 const float* __restrict__ bias, const float* __restrict__ resid,
                 float* __restrict__ outf, _Float16* __restrict__ outh,
                 int M, int N, int K)
{
  __shared__ _Float16 At[2 * BM * PA];
  __shared__ _Float16 Bt[2 * BN * PB];
  const int tid  = threadIdx.x;
  const int lane = tid & 31;
  const int wid  = tid >> 5;
  const int wm   = wid & 3;   // 4 waves along M (32 rows each)
  const int wn   = wid >> 2;  // 2 waves along N (64 cols each)
  const int m0   = blockIdx.y * BM;
  const int n0   = blockIdx.x * BN;

  v8f acc[2][4];
#pragma unroll
  for (int i = 0; i < 2; ++i)
#pragma unroll
    for (int j = 0; j < 4; ++j) acc[i][j] = (v8f){0,0,0,0,0,0,0,0};

  const int lrow = tid >> 1;            // 128 rows, 2 threads/row
  const int lofs = (tid & 1) * 16;      // 16 halves each
  // per-thread LDS destinations (low 32 bits of generic addr = LDS byte offset)
  unsigned ldsA0 = (unsigned)(size_t)(At + lrow * PA + lofs);
  unsigned ldsA1 = ldsA0 + BM * PA * 2;
  unsigned ldsB0 = (unsigned)(size_t)(Bt + lrow * PB + lofs);
  unsigned ldsB1 = ldsB0 + BN * PB * 2;
  unsigned long long ga = (unsigned long long)(A  + (long)(m0 + lrow) * K + lofs);
  unsigned long long gb = (unsigned long long)(Wt + (long)(n0 + lrow) * K + lofs);

  // prefetch tile 0 into buffer 0
  async_copy32(ldsA0, ga);
  async_copy32(ldsB0, gb);
  ga += BK * 2; gb += BK * 2;

  int p = 0;
  for (int kb = 0; kb < K; kb += BK) {
    const bool hasnext = (kb + BK < K);
    if (hasnext) {
      async_copy32(p ? ldsA0 : ldsA1, ga);
      async_copy32(p ? ldsB0 : ldsB1, gb);
      ga += BK * 2; gb += BK * 2;
      wait_async(4);
    } else {
      wait_async(0);
    }
    __syncthreads();                 // current buffer resident for all waves

    const _Float16* Ab = At + p * (BM * PA);
    const _Float16* Bb = Bt + p * (BN * PB);
    v16h af0 = ldsA(Ab, PA, wm * 32,      lane);
    v16h af1 = ldsA(Ab, PA, wm * 32 + 16, lane);
    v16h bf[4];
#pragma unroll
    for (int j = 0; j < 4; ++j) bf[j] = ldsB(Bb, PB, wn * 64 + j * 16, 0, lane);
#pragma unroll
    for (int j = 0; j < 4; ++j) {
      acc[0][j] = __builtin_amdgcn_wmma_f32_16x16x32_f16(false, af0, false, bf[j],
                                                         (short)0, acc[0][j], false, false);
      acc[1][j] = __builtin_amdgcn_wmma_f32_16x16x32_f16(false, af1, false, bf[j],
                                                         (short)0, acc[1][j], false, false);
    }
    __syncthreads();                 // all reads done before this buffer is overwritten
    p ^= 1;
  }

  const int ln15 = lane & 15, hi = lane >> 4;
#pragma unroll
  for (int j = 0; j < 4; ++j) {
    const int gn = n0 + wn * 64 + j * 16 + ln15;
    const float bj = bias[gn];
#pragma unroll
    for (int i = 0; i < 2; ++i) {
#pragma unroll
      for (int r = 0; r < 8; ++r) {
        const int gm = m0 + wm * 32 + i * 16 + r + 8 * hi;
        float v = acc[i][j][r] + bj;
        if (RELU) v = fmaxf(v, 0.0f);
        if (MODE == 0) {
          outh[(long)gm * N + gn] = (_Float16)v;
        } else if (MODE == 1) {
          const int b = gm >> 10, s = gm & 1023, h = gn >> 6, hd = gn & 63;
          outh[((long)((b * Hh + h) * Sq + s) << 6) + hd] = (_Float16)v;
        } else if (MODE == 2) {
          outf[(long)gm * N + gn] = resid[(long)gm * N + gn] + v;
        } else {
          outf[(long)gm * N + gn] = v;
        }
      }
    }
  }
}

// ---------------- flash attention ----------------------------------------
// grid (S/64, B*H), 128 threads (4 waves x 16 query rows). q,k,v: [B,H,S,HD] f16
#define KP 72
#define VP 40
#define PP 40

__global__ __launch_bounds__(128)
void attn_kernel(const _Float16* __restrict__ q, const _Float16* __restrict__ k,
                 const _Float16* __restrict__ v, _Float16* __restrict__ ctx)
{
  __shared__ _Float16 Kt[32 * KP];        // [s_k][hd]
  __shared__ _Float16 Vt[64 * VP];        // [hd][s_k]  (transposed)
  __shared__ _Float16 Pt[4 * 16 * PP];    // per-wave P tile [16][32]

  const int tid = threadIdx.x, lane = tid & 31, w = tid >> 5;
  const int bh = blockIdx.y, b = bh >> 3, h = bh & 7;
  const int q0blk = blockIdx.x * 64;
  const int qrow0 = q0blk + w * 16;
  const int r15 = lane & 15, hi = lane >> 4;

  // q A-fragments straight from global (row stride 64 halves -> 16B aligned chunks)
  const _Float16* qbase = q + ((long)bh * Sq + qrow0 + r15) * HDd;
  F16Frag qf0, qf1;
  {
    const int hi8 = hi * 8;
    qf0.u[0] = *(const uint4*)(qbase + hi8);
    qf0.u[1] = *(const uint4*)(qbase + 16 + hi8);
    qf1.u[0] = *(const uint4*)(qbase + 32 + hi8);
    qf1.u[1] = *(const uint4*)(qbase + 48 + hi8);
  }

  float m8[8], l8[8];
  v8f acc[4];
#pragma unroll
  for (int r = 0; r < 8; ++r) { m8[r] = -1e30f; l8[r] = 0.0f; }
#pragma unroll
  for (int j = 0; j < 4; ++j) acc[j] = (v8f){0,0,0,0,0,0,0,0};

  _Float16* Pw = Pt + w * 16 * PP;
  const int nkt = (q0blk + 64) >> 5;   // uniform over block -> barriers legal

  const int srow = tid >> 2;
  const int scol = (tid & 3) * 16;
  const unsigned ldsK = (unsigned)(size_t)(Kt + srow * KP + scol);

  for (int it = 0; it < nkt; ++it) {
    const int kt = it * 32;
    // ---- stage K tile (async to LDS) and transposed V tile ----
    {
      const _Float16* vs = v + ((long)bh * Sq + kt + srow) * HDd + scol;
      uint4 v0v = *(const uint4*)vs; uint4 v1v = *(const uint4*)(vs + 8);
      __syncthreads();   // previous iteration's LDS reads done
      async_copy32(ldsK, (unsigned long long)(k + ((long)bh * Sq + kt + srow) * HDd + scol));
      union { uint4 u; _Float16 e[8]; } t0, t1; t0.u = v0v; t1.u = v1v;
#pragma unroll
      for (int e = 0; e < 8; ++e) {
        Vt[(scol + e) * VP + srow]     = t0.e[e];
        Vt[(scol + 8 + e) * VP + srow] = t1.e[e];
      }
      wait_async(0);
      __syncthreads();
    }

    // ---- scores: 16q x 32k as two 16x16 accumulators ----
    v8f c0 = (v8f){0,0,0,0,0,0,0,0}, c1 = (v8f){0,0,0,0,0,0,0,0};
    c0 = __builtin_amdgcn_wmma_f32_16x16x32_f16(false, qf0.h, false, ldsB(Kt, KP, 0, 0, lane),
                                                (short)0, c0, false, false);
    c0 = __builtin_amdgcn_wmma_f32_16x16x32_f16(false, qf1.h, false, ldsB(Kt, KP, 0, 32, lane),
                                                (short)0, c0, false, false);
    c1 = __builtin_amdgcn_wmma_f32_16x16x32_f16(false, qf0.h, false, ldsB(Kt, KP, 16, 0, lane),
                                                (short)0, c1, false, false);
    c1 = __builtin_amdgcn_wmma_f32_16x16x32_f16(false, qf1.h, false, ldsB(Kt, KP, 16, 32, lane),
                                                (short)0, c1, false, false);

    // ---- online softmax update (DPP16 row reductions, no LDS) ----
    const int gk0 = kt + r15, gk1 = kt + 16 + r15;
#pragma unroll
    for (int r = 0; r < 8; ++r) {
      const int gq = qrow0 + r + 8 * hi;
      float s0 = c0[r] * 0.125f;  if (gk0 > gq) s0 = -1e30f;
      float s1 = c1[r] * 0.125f;  if (gk1 > gq) s1 = -1e30f;
      const float rm = redmax16(fmaxf(s0, s1));
      const float mn = fmaxf(m8[r], rm);
      const float alpha = __expf(m8[r] - mn);
      const float p0 = __expf(s0 - mn);
      const float p1 = __expf(s1 - mn);
      const float rs = redsum16(p0 + p1);
      l8[r] = l8[r] * alpha + rs;
      m8[r] = mn;
#pragma unroll
      for (int j = 0; j < 4; ++j) acc[j][r] *= alpha;
      Pw[(r + 8 * hi) * PP + r15]      = (_Float16)p0;
      Pw[(r + 8 * hi) * PP + 16 + r15] = (_Float16)p1;
    }
    asm volatile("s_wait_dscnt 0" ::: "memory");  // wave-private P visible

    // ---- P @ V ----
    v16h pf = ldsA(Pw, PP, 0, lane);
#pragma unroll
    for (int j = 0; j < 4; ++j)
      acc[j] = __builtin_amdgcn_wmma_f32_16x16x32_f16(false, pf, false,
                  ldsB(Vt, VP, j * 16, 0, lane), (short)0, acc[j], false, false);
  }

  // ---- normalize + write ctx [B,S,D] f16 ----
  float rl[8];
#pragma unroll
  for (int r = 0; r < 8; ++r) rl[r] = (l8[r] > 0.0f) ? 1.0f / l8[r] : 0.0f;
#pragma unroll
  for (int j = 0; j < 4; ++j) {
    const int d = h * HDd + j * 16 + r15;
#pragma unroll
    for (int r = 0; r < 8; ++r) {
      const int s = qrow0 + r + 8 * hi;
      ctx[((long)b * Sq + s) * Dm + d] = (_Float16)(acc[j][r] * rl[r]);
    }
  }
}

// ---------------- layernorm (block per row, D=512) -------------------------
__global__ __launch_bounds__(256)
void ln_kernel(const float* __restrict__ in, const float* __restrict__ g,
               const float* __restrict__ be, float* __restrict__ of,
               _Float16* __restrict__ oh)
{
  __shared__ float red[256];
  const int row = blockIdx.x, t = threadIdx.x;
  const float* p = in + (long)row * Dm;
  const float v0 = p[t], v1 = p[t + 256];
  red[t] = v0 + v1; __syncthreads();
#pragma unroll
  for (int o = 128; o > 0; o >>= 1) { if (t < o) red[t] += red[t + o]; __syncthreads(); }
  const float mean = red[0] * (1.0f / Dm);
  __syncthreads();
  const float d0 = v0 - mean, d1 = v1 - mean;
  red[t] = d0 * d0 + d1 * d1; __syncthreads();
#pragma unroll
  for (int o = 128; o > 0; o >>= 1) { if (t < o) red[t] += red[t + o]; __syncthreads(); }
  const float rstd = rsqrtf(red[0] * (1.0f / Dm) + 1e-5f);
  const float o0 = d0 * rstd * g[t] + be[t];
  const float o1 = d1 * rstd * g[t + 256] + be[t + 256];
  if (of) { of[(long)row * Dm + t] = o0; of[(long)row * Dm + t + 256] = o1; }
  oh[(long)row * Dm + t]       = (_Float16)o0;
  oh[(long)row * Dm + t + 256] = (_Float16)o1;
}

// ---------------- embedding + positional encoding --------------------------
__global__ __launch_bounds__(256)
void embed_kernel(const int* __restrict__ tokens, const float* __restrict__ emb,
                  const float* __restrict__ pe, float* __restrict__ x,
                  _Float16* __restrict__ xh)
{
  const long i = (long)blockIdx.x * 256 + threadIdx.x;  // B*S*D total
  const int d = (int)(i % Dm);
  const long sd = i / Dm;
  const int s = (int)(sd % Sq);
  const int b = (int)(sd / Sq);
  const float val = emb[(long)tokens[b * Sq + s] * Dm + d] + pe[s * Dm + d];
  x[i] = val;
  xh[i] = (_Float16)val;
}

// ---------------- weight transpose-convert f32[K][N] -> f16[N][K] ----------
__global__ __launch_bounds__(256)
void convw_kernel(const float* __restrict__ w, _Float16* __restrict__ wt, int K, int N)
{
  __shared__ float tile[32][33];
  const int tx = threadIdx.x & 31, ty = threadIdx.x >> 5;
  const int nb = blockIdx.x * 32, kb = blockIdx.y * 32;
#pragma unroll
  for (int i = 0; i < 4; ++i) {
    const int kk = ty + i * 8;
    tile[kk][tx] = w[(long)(kb + kk) * N + nb + tx];
  }
  __syncthreads();
#pragma unroll
  for (int i = 0; i < 4; ++i) {
    const int nn = ty + i * 8;
    wt[(long)(nb + nn) * K + kb + tx] = (_Float16)tile[tx][nn];
  }
}

// ---------------- host orchestration ---------------------------------------
extern "C" void kernel_launch(void* const* d_in, const int* in_sizes, int n_in,
                              void* d_out, int out_size, void* d_ws, size_t ws_size,
                              hipStream_t stream)
{
  (void)in_sizes; (void)n_in; (void)out_size; (void)ws_size;
  const int*   tokens = (const int*)  d_in[0];
  const float* pe     = (const float*)d_in[2];
  const float* emb    = (const float*)d_in[3];
  const float* Wq = (const float*)d_in[4];  const float* bq = (const float*)d_in[5];
  const float* Wk = (const float*)d_in[6];  const float* bk = (const float*)d_in[7];
  const float* Wv = (const float*)d_in[8];  const float* bv = (const float*)d_in[9];
  const float* Wo = (const float*)d_in[10]; const float* bo = (const float*)d_in[11];
  const float* l1s = (const float*)d_in[12]; const float* l1b = (const float*)d_in[13];
  const float* W1 = (const float*)d_in[14]; const float* b1 = (const float*)d_in[15];
  const float* W2 = (const float*)d_in[16]; const float* b2 = (const float*)d_in[17];
  const float* l2s = (const float*)d_in[18]; const float* l2b = (const float*)d_in[19];
  const float* lfs = (const float*)d_in[20]; const float* lfb = (const float*)d_in[21];
  const float* Wout = (const float*)d_in[22]; const float* bout = (const float*)d_in[23];
  float* out = (float*)d_out;

  const long MT = (long)Bsz * Sq;   // 2048 token rows
  char* ws = (char*)d_ws;
  auto carve = [&](size_t bytes) { char* p = ws; ws += (bytes + 255) & ~(size_t)255; return p; };

  float*    x    = (float*)   carve(MT * Dm * 4);
  float*    y    = (float*)   carve(MT * Dm * 4);
  _Float16* xh   = (_Float16*)carve(MT * Dm * 2);
  _Float16* qbuf = (_Float16*)carve(MT * Dm * 2);
  _Float16* kbuf = (_Float16*)carve(MT * Dm * 2);
  _Float16* vbuf = (_Float16*)carve(MT * Dm * 2);
  _Float16* ctx  = (_Float16*)carve(MT * Dm * 2);
  _Float16* h1   = (_Float16*)carve(MT * FFd * 2);
  _Float16* wslot= (_Float16*)carve((size_t)Vv * Dm * 2);   // reused per layer + Wout

  _Float16* wq_t = wslot;
  _Float16* wk_t = wslot + 1 * Dm * Dm;
  _Float16* wv_t = wslot + 2 * Dm * Dm;
  _Float16* wo_t = wslot + 3 * Dm * Dm;
  _Float16* w1_t = wslot + 4 * Dm * Dm;              // [FF][D]
  _Float16* w2_t = w1_t + (long)FFd * Dm;            // [D][FF]

  embed_kernel<<<(int)(MT * Dm / 256), 256, 0, stream>>>(tokens, emb, pe, x, xh);

  const dim3 gDD(Dm / BN, MT / BM);     // (4,16)
  const dim3 gDF(FFd / BN, MT / BM);    // (16,16)
  const dim3 gAtt(Sq / 64, Bsz * Hh);   // (16,16)

  for (int l = 0; l < Ll; ++l) {
    convw_kernel<<<dim3(Dm / 32, Dm / 32), 256, 0, stream>>>(Wq + (long)l * Dm * Dm, wq_t, Dm, Dm);
    convw_kernel<<<dim3(Dm / 32, Dm / 32), 256, 0, stream>>>(Wk + (long)l * Dm * Dm, wk_t, Dm, Dm);
    convw_kernel<<<dim3(Dm / 32, Dm / 32), 256, 0, stream>>>(Wv + (long)l * Dm * Dm, wv_t, Dm, Dm);
    convw_kernel<<<dim3(Dm / 32, Dm / 32), 256, 0, stream>>>(Wo + (long)l * Dm * Dm, wo_t, Dm, Dm);
    convw_kernel<<<dim3(FFd / 32, Dm / 32), 256, 0, stream>>>(W1 + (long)l * Dm * FFd, w1_t, Dm, FFd);
    convw_kernel<<<dim3(Dm / 32, FFd / 32), 256, 0, stream>>>(W2 + (long)l * FFd * Dm, w2_t, FFd, Dm);

    gemm_kernel<1, false><<<gDD, 256, 0, stream>>>(xh, wq_t, bq + (long)l * Dm, nullptr,
                                                   nullptr, qbuf, (int)MT, Dm, Dm);
    gemm_kernel<1, false><<<gDD, 256, 0, stream>>>(xh, wk_t, bk + (long)l * Dm, nullptr,
                                                   nullptr, kbuf, (int)MT, Dm, Dm);
    gemm_kernel<1, false><<<gDD, 256, 0, stream>>>(xh, wv_t, bv + (long)l * Dm, nullptr,
                                                   nullptr, vbuf, (int)MT, Dm, Dm);

    attn_kernel<<<gAtt, 128, 0, stream>>>(qbuf, kbuf, vbuf, ctx);

    gemm_kernel<2, false><<<gDD, 256, 0, stream>>>(ctx, wo_t, bo + (long)l * Dm, x,
                                                   y, nullptr, (int)MT, Dm, Dm);
    ln_kernel<<<(int)MT, 256, 0, stream>>>(y, l1s + (long)l * Dm, l1b + (long)l * Dm, x, xh);

    gemm_kernel<0, true><<<gDF, 256, 0, stream>>>(xh, w1_t, b1 + (long)l * FFd, nullptr,
                                                  nullptr, h1, (int)MT, FFd, Dm);
    gemm_kernel<2, false><<<gDD, 256, 0, stream>>>(h1, w2_t, b2 + (long)l * Dm, x,
                                                   y, nullptr, (int)MT, Dm, FFd);
    ln_kernel<<<(int)MT, 256, 0, stream>>>(y, l2s + (long)l * Dm, l2b + (long)l * Dm, x, xh);
  }

  ln_kernel<<<(int)MT, 256, 0, stream>>>(x, lfs, lfb, nullptr, xh);
  convw_kernel<<<dim3(Vv / 32, Dm / 32), 256, 0, stream>>>(Wout, wslot, Dm, Vv);
  gemm_kernel<3, false><<<dim3(Vv / BN, MT / BM), 256, 0, stream>>>(xh, wslot, bout, nullptr,
                                                                    out, nullptr, (int)MT, Vv, Dm);
}
